// _RelPN_62938450755908
// MI455X (gfx1250) — matrix operand order repaired and checked by reading, not compiled, not run on hardware
//
#include <hip/hip_runtime.h>
#include <hip/hip_bf16.h>

typedef __attribute__((ext_vector_type(16))) _Float16 v16h;
typedef __attribute__((ext_vector_type(8)))  _Float16 v8h;
typedef __attribute__((ext_vector_type(8)))  float    v8f;

#define BATCH 8
#define NROI  2048
#define CFEAT 151
#define DIN   155
#define HID   64
#define KSEL  2048
#define EQCAP 8192
#define NBIN1 8192   // key >> 17   (monotone; score<=1.0 -> bin<=8128)
#define NBIN2 2048   // (key >> 6) & 2047
#define NBIN3 64     // key & 63

// ---------------- workspace layout (bytes) ----------------
static constexpr size_t OFF_XS  = 0;
static constexpr size_t OFF_XO  = OFF_XS + (size_t)BATCH*NROI*HID*2;
static constexpr size_t OFF_H1  = OFF_XO + (size_t)BATCH*NROI*HID*2;
static constexpr size_t OFF_H2  = OFF_H1 + (size_t)BATCH*NBIN1*4;
static constexpr size_t OFF_H3  = OFF_H2 + (size_t)BATCH*NBIN2*4;
static constexpr size_t OFF_SEL = OFF_H3 + (size_t)BATCH*NBIN3*4;
static constexpr size_t OFF_GT  = OFF_SEL + (size_t)BATCH*16*4;
static constexpr size_t OFF_EQ  = OFF_GT + (size_t)BATCH*KSEL*8;
static constexpr size_t ZERO_WORDS = (OFF_GT - OFF_H1) / 4;   // hists + sel

// ---------------- 0: zero histograms / counters ----------------
__global__ void k_init(unsigned* __restrict__ z, unsigned nwords) {
  for (unsigned i = blockIdx.x * blockDim.x + threadIdx.x; i < nwords;
       i += gridDim.x * blockDim.x)
    z[i] = 0u;
}

// ---------------- 1: fused feat + two MLPs -> f16 activations ----------------
__global__ __launch_bounds__(256) void k_mlp(
    const float* __restrict__ rois, const float* __restrict__ roi_feat,
    const float* __restrict__ im_info,
    const float* __restrict__ w1s, const float* __restrict__ b1s,
    const float* __restrict__ w2s, const float* __restrict__ b2s,
    const float* __restrict__ w1o, const float* __restrict__ b1o,
    const float* __restrict__ w2o, const float* __restrict__ b2o,
    _Float16* __restrict__ xs, _Float16* __restrict__ xo) {
  __shared__ float feat[4][160];
  __shared__ float hs_[4][HID];
  __shared__ float ho_[4][HID];
  const int tid = threadIdx.x;
  const int r   = tid >> 6;
  const int j   = tid & 63;
  const int gr  = blockIdx.x * 4 + r;    // global row in [0, B*N)
  const int b   = gr >> 11;
  const float* rf = roi_feat + (size_t)gr * CFEAT;
  for (int d = j; d < CFEAT; d += 64) feat[r][d] = rf[d];
  if (j == 0) {
    const float* ro = rois + (size_t)gr * 5;
    const float sxy = im_info[b * 3 + 1];
    const float swh = im_info[b * 3 + 0];
    feat[r][151] = ro[1] / sxy;
    feat[r][152] = ro[2] / sxy;
    feat[r][153] = ro[3] / swh;
    feat[r][154] = ro[4] / swh;
  }
  __syncthreads();
  float as = b1s[j], ao = b1o[j];
  for (int d = 0; d < DIN; ++d) {
    const float f = feat[r][d];
    as = fmaf(f, w1s[d * HID + j], as);
    ao = fmaf(f, w1o[d * HID + j], ao);
  }
  hs_[r][j] = fmaxf(as, 0.0f);
  ho_[r][j] = fmaxf(ao, 0.0f);
  __syncthreads();
  float os = b2s[j], oo = b2o[j];
  for (int d = 0; d < HID; ++d) {
    os = fmaf(hs_[r][d], w2s[d * HID + j], os);
    oo = fmaf(ho_[r][d], w2o[d * HID + j], oo);
  }
  xs[(size_t)gr * HID + j] = (_Float16)os;
  xo[(size_t)gr * HID + j] = (_Float16)oo;
}

// ---------------- 2: WMMA bilinear + sigmoid + level-1 histogram ----------------
// grid: (N/512, N/16, B), block 256 (8 waves); each wave -> 16x64 strip (4 tiles)
__global__ __launch_bounds__(256) void k_bilinear(
    const _Float16* __restrict__ xs, const _Float16* __restrict__ xo,
    float* __restrict__ vis, unsigned* __restrict__ hist1) {
  __shared__ unsigned lh[NBIN1];           // 32 KB privatized histogram
  const int tid = threadIdx.x;
  for (int i = tid; i < NBIN1; i += 256) lh[i] = 0u;
  __syncthreads();
  const int b    = blockIdx.z;
  const int wave = tid >> 5;
  const int lane = tid & 31;
  const int n0 = blockIdx.y * 16;
  const int m0 = (blockIdx.x * 8 + wave) * 64;   // 4 consecutive 16-col tiles
  const int hs = lane >> 4;
  const int rl = lane & 15;
  const _Float16* arow = xs + ((size_t)b * NROI + n0 + rl) * HID;
  const _Float16* bbase = xo + ((size_t)b * NROI + rl) * HID;
  v8f c0 = {}, c1 = {}, c2 = {}, c3 = {};
  v8f* acc[4] = {&c0, &c1, &c2, &c3};
#pragma unroll
  for (int k0 = 0; k0 < 64; k0 += 32) {
    // Fetch ALL fragments for this K-step first: one load clause, one wait,
    // then 4 independent back-to-back WMMAs (no RAW between accumulators).
    // A 16x32 f16: lane<16 holds K[0..7],K[16..23]; lane>=16 holds K[8..15],K[24..31]
    const v8h alo = *(const v8h*)(arow + k0 + 8 * hs);
    const v8h ahi = *(const v8h*)(arow + k0 + 16 + 8 * hs);
    v8h blo[4], bhi[4];
#pragma unroll
    for (int t = 0; t < 4; ++t) {
      // B 32x16 f16: lane holds col rl of tile t, K = 8*hs + 0..7, 16 + 8*hs + 0..7 (same as A)
      const _Float16* brow = bbase + (size_t)(m0 + t * 16) * HID;
      blo[t] = *(const v8h*)(brow + k0 + 8 * hs);
      bhi[t] = *(const v8h*)(brow + k0 + 16 + 8 * hs);
    }
    v16h A;
#pragma unroll
    for (int i = 0; i < 8; ++i) { A[i] = alo[i]; A[8 + i] = ahi[i]; }
#pragma unroll
    for (int t = 0; t < 4; ++t) {
      v16h Bm;
#pragma unroll
      for (int i = 0; i < 8; ++i) { Bm[i] = blo[t][i]; Bm[8 + i] = bhi[t][i]; }
      *acc[t] = __builtin_amdgcn_wmma_f32_16x16x32_f16(false, A, false, Bm,
                                                       (short)0, *acc[t], false, false);
    }
  }
  float* outb = vis + (size_t)b * NROI * NROI;
  const v8f cc_[4] = {c0, c1, c2, c3};
#pragma unroll
  for (int t = 0; t < 4; ++t) {
#pragma unroll
    for (int i = 0; i < 8; ++i) {
      const int rr = n0 + i + 8 * hs;          // D: VGPR i -> M = i + 8*(lane/16)
      const int cc = m0 + t * 16 + rl;
      const float e = __expf(-cc_[t][i]);
      const float s = __builtin_amdgcn_rcpf(1.0f + e);   // fast sigmoid
      outb[(size_t)rr * NROI + cc] = s;
      atomicAdd(&lh[__float_as_uint(s) >> 17], 1u);
    }
  }
  __syncthreads();
  unsigned* gh = hist1 + b * NBIN1;
  for (int i = tid; i < NBIN1; i += 256)
    if (lh[i]) atomicAdd(&gh[i], lh[i]);
}

// ---------------- serial suffix scans (1 block per batch; trivial cost) ----------------
__global__ void k_scan1(const unsigned* __restrict__ h1, unsigned* __restrict__ sel) {
  if (threadIdx.x) return;
  const int b = blockIdx.x;
  const unsigned* h = h1 + b * NBIN1;
  unsigned cum = 0;
  for (int bin = NBIN1 - 1; bin >= 0; --bin) {
    const unsigned c = h[bin];
    if (cum + c >= (unsigned)KSEL) { sel[b*16+0] = bin; sel[b*16+1] = cum; return; }
    cum += c;
  }
  sel[b*16+0] = 0; sel[b*16+1] = cum;
}

__global__ void k_scan2(const unsigned* __restrict__ h2, unsigned* __restrict__ sel) {
  if (threadIdx.x) return;
  const int b = blockIdx.x;
  const unsigned* h = h2 + b * NBIN2;
  unsigned cum = sel[b*16+1];
  for (int bin = NBIN2 - 1; bin >= 0; --bin) {
    const unsigned c = h[bin];
    if (cum + c >= (unsigned)KSEL) { sel[b*16+2] = bin; sel[b*16+3] = cum; return; }
    cum += c;
  }
  sel[b*16+2] = 0; sel[b*16+3] = cum;
}

__global__ void k_scan3(const unsigned* __restrict__ h3, unsigned* __restrict__ sel) {
  if (threadIdx.x) return;
  const int b = blockIdx.x;
  const unsigned* h = h3 + b * NBIN3;
  const unsigned bin1 = sel[b*16+0], bin2 = sel[b*16+2];
  unsigned cum = sel[b*16+3];
  for (int bin = NBIN3 - 1; bin >= 0; --bin) {
    const unsigned c = h[bin];
    if (cum + c >= (unsigned)KSEL) {
      sel[b*16+4] = (bin1 << 17) | (bin2 << 6) | (unsigned)bin;  // exact K-th key T
      sel[b*16+5] = cum;                                         // count of keys > T
      return;
    }
    cum += c;
  }
  sel[b*16+4] = 0; sel[b*16+5] = cum;
}

// ---------------- filtered refinement histograms (b128 reads) ----------------
__global__ void k_hist2(const float* __restrict__ vis, const unsigned* __restrict__ sel,
                        unsigned* __restrict__ h2) {
  const int b = blockIdx.y;
  const unsigned bin1 = sel[b*16+0];
  const float4* v = (const float4*)(vis + (size_t)b * NROI * NROI);
  const int total4 = NROI * NROI / 4;
  for (int i = blockIdx.x * blockDim.x + threadIdx.x; i < total4;
       i += gridDim.x * blockDim.x) {
    const float4 f = v[i];
    const float e[4] = {f.x, f.y, f.z, f.w};
#pragma unroll
    for (int c = 0; c < 4; ++c) {
      const unsigned key = __float_as_uint(e[c]);
      if ((key >> 17) == bin1) atomicAdd(&h2[b * NBIN2 + ((key >> 6) & 2047)], 1u);
    }
  }
}

__global__ void k_hist3(const float* __restrict__ vis, const unsigned* __restrict__ sel,
                        unsigned* __restrict__ h3) {
  const int b = blockIdx.y;
  const unsigned bin1 = sel[b*16+0], bin2 = sel[b*16+2];
  const unsigned pre = (bin1 << 11) | bin2;   // bits 30..6 of matching keys
  const float4* v = (const float4*)(vis + (size_t)b * NROI * NROI);
  const int total4 = NROI * NROI / 4;
  for (int i = blockIdx.x * blockDim.x + threadIdx.x; i < total4;
       i += gridDim.x * blockDim.x) {
    const float4 f = v[i];
    const float e[4] = {f.x, f.y, f.z, f.w};
#pragma unroll
    for (int c = 0; c < 4; ++c) {
      const unsigned key = __float_as_uint(e[c]);
      if ((key >> 6) == pre) atomicAdd(&h3[b * NBIN3 + (key & 63)], 1u);
    }
  }
}

// ---------------- compact ">T" and "==T" sets (b128 reads) ----------------
__global__ void k_compact(const float* __restrict__ vis, unsigned* __restrict__ sel,
                          uint2* __restrict__ gt, unsigned* __restrict__ eq) {
  const int b = blockIdx.y;
  const unsigned T = sel[b*16+4];
  const float4* v = (const float4*)(vis + (size_t)b * NROI * NROI);
  const int total4 = NROI * NROI / 4;
  for (int i = blockIdx.x * blockDim.x + threadIdx.x; i < total4;
       i += gridDim.x * blockDim.x) {
    const float4 f = v[i];
    const float e[4] = {f.x, f.y, f.z, f.w};
#pragma unroll
    for (int c = 0; c < 4; ++c) {
      const unsigned key = __float_as_uint(e[c]);
      if (key > T) {
        const unsigned p = atomicAdd(&sel[b*16+7], 1u);
        if (p < (unsigned)KSEL) gt[b * KSEL + p] = make_uint2(key, (unsigned)(i * 4 + c));
      } else if (key == T) {
        const unsigned p = atomicAdd(&sel[b*16+8], 1u);
        if (p < (unsigned)EQCAP) eq[b * EQCAP + p] = (unsigned)(i * 4 + c);
      }
    }
  }
}

// ---------------- per-batch sort + emit outputs ----------------
__global__ __launch_bounds__(1024) void k_final(
    const uint2* __restrict__ gt, const unsigned* __restrict__ eq,
    const unsigned* __restrict__ sel, const float* __restrict__ rois,
    float* __restrict__ roi_pairs, float* __restrict__ roi_prop,
    float* __restrict__ top_scores) {
  __shared__ unsigned skey[KSEL];
  __shared__ unsigned sidx[KSEL];
  __shared__ unsigned eidx[EQCAP];
  const int b = blockIdx.x, tid = threadIdx.x;
  const unsigned C = sel[b*16+5];
  const unsigned T = sel[b*16+4];
  const unsigned gcnt = sel[b*16+7] < (unsigned)KSEL  ? sel[b*16+7] : (unsigned)KSEL;
  const unsigned ecnt = sel[b*16+8] < (unsigned)EQCAP ? sel[b*16+8] : (unsigned)EQCAP;
  for (int i = tid; i < KSEL; i += 1024) {
    if (i < (int)gcnt) { const uint2 g = gt[b * KSEL + i]; skey[i] = g.x; sidx[i] = g.y; }
    else               { skey[i] = 0u; sidx[i] = 0xFFFFFFFFu; }
  }
  for (int i = tid; i < EQCAP; i += 1024)
    eidx[i] = (i < (int)ecnt) ? eq[b * EQCAP + i] : 0xFFFFFFFFu;
  __syncthreads();
  // bitonic sort (key desc, idx asc) on 2048 pairs
  for (unsigned k = 2; k <= (unsigned)KSEL; k <<= 1)
    for (unsigned j = k >> 1; j > 0; j >>= 1) {
      for (unsigned i = tid; i < (unsigned)KSEL; i += 1024) {
        const unsigned ixj = i ^ j;
        if (ixj > i) {
          const bool up = ((i & k) == 0);
          const unsigned ka = skey[i], kb = skey[ixj];
          const unsigned ia = sidx[i], ib = sidx[ixj];
          const bool ord = (ka > kb) || (ka == kb && ia < ib);
          if (up != ord) { skey[i] = kb; skey[ixj] = ka; sidx[i] = ib; sidx[ixj] = ia; }
        }
      }
      __syncthreads();
    }
  // bitonic sort eq indices ascending (ties at T break by lowest index)
  for (unsigned k = 2; k <= (unsigned)EQCAP; k <<= 1)
    for (unsigned j = k >> 1; j > 0; j >>= 1) {
      for (unsigned i = tid; i < (unsigned)EQCAP; i += 1024) {
        const unsigned ixj = i ^ j;
        if (ixj > i) {
          const bool up = ((i & k) == 0);
          const unsigned a = eidx[i], c2 = eidx[ixj];
          if (up != (a <= c2)) { eidx[i] = c2; eidx[ixj] = a; }
        }
      }
      __syncthreads();
    }
  for (int r = tid; r < KSEL; r += 1024) {
    unsigned idx; float score;
    if (r < (int)C) { idx = sidx[r]; score = __uint_as_float(skey[r]); }
    else {
      const unsigned e = (unsigned)r - C;
      idx = (e < (unsigned)EQCAP) ? eidx[e] : 0u;
      score = __uint_as_float(T);
    }
    if (idx == 0xFFFFFFFFu) idx = 0u;
    const unsigned sub = idx >> 11, obj = idx & 2047;
    top_scores[(size_t)b * KSEL + r] = score;
    roi_prop[((size_t)b * KSEL + r) * 2 + 0] = (float)sub;
    roi_prop[((size_t)b * KSEL + r) * 2 + 1] = (float)obj;
    const float* rs = rois + ((size_t)b * NROI + sub) * 5;
    const float* ro = rois + ((size_t)b * NROI + obj) * 5;
    float* rp = roi_pairs + ((size_t)b * KSEL + r) * 8;
    rp[0] = rs[1]; rp[1] = rs[2]; rp[2] = rs[3]; rp[3] = rs[4];
    rp[4] = ro[1]; rp[5] = ro[2]; rp[6] = ro[3]; rp[7] = ro[4];
  }
}

extern "C" void kernel_launch(void* const* d_in, const int* in_sizes, int n_in,
                              void* d_out, int out_size, void* d_ws, size_t ws_size,
                              hipStream_t stream) {
  const float* rois     = (const float*)d_in[0];
  const float* roi_feat = (const float*)d_in[1];
  const float* im_info  = (const float*)d_in[2];
  const float* w1s = (const float*)d_in[3];
  const float* b1s = (const float*)d_in[4];
  const float* w2s = (const float*)d_in[5];
  const float* b2s = (const float*)d_in[6];
  const float* w1o = (const float*)d_in[7];
  const float* b1o = (const float*)d_in[8];
  const float* w2o = (const float*)d_in[9];
  const float* b2o = (const float*)d_in[10];

  float* out        = (float*)d_out;
  float* roi_pairs  = out;                      // [8,2048,8]
  float* roi_prop   = out + 131072;             // [8,2048,2]
  float* top_scores = out + 163840;             // [8,2048]
  float* vis        = out + 180224;             // [8, 2048*2048]

  char* ws = (char*)d_ws;
  _Float16* xs = (_Float16*)(ws + OFF_XS);
  _Float16* xo = (_Float16*)(ws + OFF_XO);
  unsigned* h1 = (unsigned*)(ws + OFF_H1);
  unsigned* h2 = (unsigned*)(ws + OFF_H2);
  unsigned* h3 = (unsigned*)(ws + OFF_H3);
  unsigned* sel = (unsigned*)(ws + OFF_SEL);
  uint2*    gt  = (uint2*)(ws + OFF_GT);
  unsigned* eq  = (unsigned*)(ws + OFF_EQ);

  k_init<<<256, 256, 0, stream>>>(h1, (unsigned)ZERO_WORDS);
  k_mlp<<<BATCH * NROI / 4, 256, 0, stream>>>(rois, roi_feat, im_info,
      w1s, b1s, w2s, b2s, w1o, b1o, w2o, b2o, xs, xo);
  k_bilinear<<<dim3(NROI / 512, NROI / 16, BATCH), 256, 0, stream>>>(xs, xo, vis, h1);
  k_scan1<<<BATCH, 32, 0, stream>>>(h1, sel);
  k_hist2<<<dim3(512, BATCH), 256, 0, stream>>>(vis, sel, h2);
  k_scan2<<<BATCH, 32, 0, stream>>>(h2, sel);
  k_hist3<<<dim3(512, BATCH), 256, 0, stream>>>(vis, sel, h3);
  k_scan3<<<BATCH, 32, 0, stream>>>(h3, sel);
  k_compact<<<dim3(512, BATCH), 256, 0, stream>>>(vis, sel, gt, eq);
  k_final<<<BATCH, 1024, 0, stream>>>(gt, eq, sel, rois, roi_pairs, roi_prop, top_scores);
}
